// WindowedDinoV2Block_79903571574928
// MI455X (gfx1250) — compile-verified, matrix operand order checked by reading, not run on hardware
//
#include <hip/hip_runtime.h>
#include <hip/hip_bf16.h>

#define B_  32
#define S_  576
#define D_  1024
#define H_  16
#define HD_ 64
#define DFF_ 4096
#define M_  (B_ * S_)     // 18432 rows

typedef __attribute__((ext_vector_type(16))) __bf16 v16bf;
typedef __attribute__((ext_vector_type(8)))  __bf16 v8bf;
typedef __attribute__((ext_vector_type(8)))  float  v8f;

union V16U { v16bf v; v8bf h[2]; };

// Load a 16x16x32 bf16 WMMA A/B fragment from a row-major LDS tile.
// Lane L: row = rowSel; holds K = lo..lo+7 (VGPR0-3) and K = lo+16..lo+23 (VGPR4-7),
// where lo includes the per-lane base (0 for lanes 0-15, 8 for lanes 16-31).
__device__ __forceinline__ v16bf load_frag(const __bf16* base, int stride, int row, int lo) {
    V16U u;
    u.h[0] = *(const v8bf*)(base + (size_t)row * stride + lo);
    u.h[1] = *(const v8bf*)(base + (size_t)row * stride + lo + 16);
    return u.v;
}

__device__ __forceinline__ v8f wmma_bf16(v16bf a, v16bf b, v8f c) {
    return __builtin_amdgcn_wmma_f32_16x16x32_bf16(false, a, false, b, (short)0, c, false, false);
}

// Async VMEM -> LDS copy of 16 bytes per lane (CDNA5 GLOBAL_LOAD_ASYNC_TO_LDS_B128, ASYNCcnt).
__device__ __forceinline__ void async_copy16(unsigned int lds_off, const __bf16* gaddr) {
    asm volatile("global_load_async_to_lds_b128 %0, %1, off"
                 :: "v"(lds_off), "v"(gaddr) : "memory");
}
__device__ __forceinline__ void wait_async0() {
    asm volatile("s_wait_asynccnt 0" ::: "memory");
}
__device__ __forceinline__ unsigned int lds_addr32(const void* p) {
    return (unsigned int)(size_t)p;   // low 32 bits of generic address = LDS offset
}

// ---------------- LayerNorm (f32 in -> bf16 out), one row per block ----------------
__global__ __launch_bounds__(256)
void ln_bf16_kernel(const float* __restrict__ x, const float* __restrict__ g,
                    const float* __restrict__ bta, __bf16* __restrict__ out) {
    int row = blockIdx.x, t = threadIdx.x;
    const float* xr = x + (size_t)row * D_;
    float v[4];
#pragma unroll
    for (int i = 0; i < 4; i++) v[i] = xr[t + i * 256];
    float s = v[0] + v[1] + v[2] + v[3];
    __shared__ float red[8];
    for (int o = 16; o >= 1; o >>= 1) s += __shfl_xor(s, o, 32);
    if ((t & 31) == 0) red[t >> 5] = s;
    __syncthreads();
    float mean = 0.f;
#pragma unroll
    for (int i = 0; i < 8; i++) mean += red[i];
    mean *= (1.0f / (float)D_);
    float q = 0.f;
#pragma unroll
    for (int i = 0; i < 4; i++) { float d = v[i] - mean; q += d * d; }
    for (int o = 16; o >= 1; o >>= 1) q += __shfl_xor(q, o, 32);
    __syncthreads();
    if ((t & 31) == 0) red[t >> 5] = q;
    __syncthreads();
    float var = 0.f;
#pragma unroll
    for (int i = 0; i < 8; i++) var += red[i];
    var *= (1.0f / (float)D_);
    float rstd = rsqrtf(var + 1e-6f);
#pragma unroll
    for (int i = 0; i < 4; i++) {
        int c = t + i * 256;
        out[(size_t)row * D_ + c] = (__bf16)((v[i] - mean) * rstd * g[c] + bta[c]);
    }
}

// ---------------- transpose + f32 -> bf16: out[N,K] = (bf16) in[K,N] ----------------
__global__ __launch_bounds__(256)
void transpose_bf16_kernel(const float* __restrict__ in, __bf16* __restrict__ out, int K, int N) {
    __shared__ float tile[32][33];
    int n0 = blockIdx.x * 32, k0 = blockIdx.y * 32;
    int tx = threadIdx.x & 31, ty = threadIdx.x >> 5;   // 32 x 8
#pragma unroll
    for (int i = 0; i < 32; i += 8)
        tile[ty + i][tx] = in[(size_t)(k0 + ty + i) * N + n0 + tx];
    __syncthreads();
#pragma unroll
    for (int i = 0; i < 32; i += 8)
        out[(size_t)(n0 + ty + i) * K + k0 + tx] = (__bf16)tile[tx][ty + i];
}

// ---------------- Tiled bf16 WMMA GEMM: out = epilogue(A[M,K] @ Wt[N,K]^T + bias) ----------------
// Workgroup tile 256x128, 8 waves, each wave a 64x64 tile (4x4 WMMA frags, 16 wmma / K-step).
// Double-buffered LDS, staged with async VMEM->LDS copies (ASYNCcnt).
// mode 0: bf16 out = result            (QKV projections)
// mode 1: f32  out = result*ls + res   (O-proj / MLP2 with residual)
// mode 2: bf16 out = gelu(result)      (MLP1)
__global__ __launch_bounds__(256)
void gemm_bf16_kernel(const __bf16* __restrict__ A, const __bf16* __restrict__ Wt,
                      const float* __restrict__ bias, const float* __restrict__ ls,
                      const float* __restrict__ res, void* __restrict__ out,
                      int M, int N, int K, int mode) {
    __shared__ __bf16 Abuf[2][256][40];   // 256 x 32 (+8 pad) per buffer
    __shared__ __bf16 Wbuf[2][128][40];   // 128 x 32 (+8 pad) per buffer
    int n0 = blockIdx.x * 128, m0 = blockIdx.y * 256;
    int t = threadIdx.x, wave = t >> 5, lane = t & 31;
    int lh = lane & 15;
    int base  = (lane < 16) ? 0 : 8;
    int mhalf = (lane < 16) ? 0 : 8;
    int mbase = (wave >> 1) * 64;     // 4 wave-rows of 64
    int nbase = (wave & 1) * 64;      // 2 wave-cols of 64

    v8f acc[4][4] = {};

    auto stage = [&](int bufi, int k0) {
#pragma unroll
        for (int i = 0; i < 4; i++) {           // A: 256 rows x 32 cols = 1024 x 16B
            int fa = t + i * 256;
            int row = fa >> 2, c0 = (fa & 3) * 8;
            async_copy16(lds_addr32(&Abuf[bufi][row][c0]),
                         A + (size_t)(m0 + row) * K + k0 + c0);
        }
#pragma unroll
        for (int i = 0; i < 2; i++) {           // Wt: 128 rows x 32 cols = 512 x 16B
            int fw = t + i * 256;
            int row = fw >> 2, c0 = (fw & 3) * 8;
            async_copy16(lds_addr32(&Wbuf[bufi][row][c0]),
                         Wt + (size_t)(n0 + row) * K + k0 + c0);
        }
    };

    stage(0, 0);
    int nk = K >> 5;
    for (int ki = 0; ki < nk; ki++) {
        int cur = ki & 1;
        wait_async0();          // this wave's async copies into buf[cur] have landed
        __syncthreads();        // everyone's landed; prior reads of buf[cur^1] done
        if (ki + 1 < nk) stage(cur ^ 1, (ki + 1) << 5);

        v16bf af[4];
#pragma unroll
        for (int s = 0; s < 4; s++)
            af[s] = load_frag(&Abuf[cur][0][0], 40, mbase + s * 16 + lh, base);
#pragma unroll
        for (int f = 0; f < 4; f++) {
            v16bf bf = load_frag(&Wbuf[cur][0][0], 40, nbase + f * 16 + lh, base);
#pragma unroll
            for (int s = 0; s < 4; s++)
                acc[s][f] = wmma_bf16(af[s], bf, acc[s][f]);
        }
    }

#pragma unroll
    for (int f = 0; f < 4; f++) {
        int col = n0 + nbase + f * 16 + lh;
        float bv = bias[col];
        float lsv = (mode == 1) ? ls[col] : 0.f;
#pragma unroll
        for (int s = 0; s < 4; s++) {
#pragma unroll
            for (int r = 0; r < 8; r++) {
                int row = m0 + mbase + s * 16 + mhalf + r;
                size_t idx = (size_t)row * N + col;
                float v = acc[s][f][r] + bv;
                if (mode == 0) {
                    ((__bf16*)out)[idx] = (__bf16)v;
                } else if (mode == 1) {
                    ((float*)out)[idx] = v * lsv + res[idx];
                } else {
                    float gl = 0.5f * v * (1.0f + erff(v * 0.70710678118f));
                    ((__bf16*)out)[idx] = (__bf16)gl;
                }
            }
        }
    }
}

// ---------------- Flash attention: 4 waves/block share one (b,h); wave = 16 query rows ----------------
__global__ __launch_bounds__(128)
void attn_kernel(const __bf16* __restrict__ Q, const __bf16* __restrict__ Km,
                 const __bf16* __restrict__ Vm, __bf16* __restrict__ Ctx) {
    __shared__ __bf16 Ksub[32][72];       // 32 key rows x 64 hd (+8 pad)
    __shared__ __bf16 Vt[64][40];         // V transposed: 64 hd x 32 keys (+8 pad)
    __shared__ __bf16 Pscr[4][16][40];    // per-wave P re-layout scratch

    int blk = blockIdx.x;
    int b = blk / (H_ * 9);
    int rem = blk % (H_ * 9);
    int h = rem / 9, qg = rem % 9;
    int t = threadIdx.x, wave = t >> 5, lane = t & 31;
    int lh = lane & 15;
    int base  = (lane < 16) ? 0 : 8;
    int mhalf = (lane < 16) ? 0 : 8;
    int qrow0 = qg * 64 + wave * 16;

    const __bf16* Qb = Q  + (size_t)b * S_ * D_ + h * HD_;
    const __bf16* Kb = Km + (size_t)b * S_ * D_ + h * HD_;
    const __bf16* Vb = Vm + (size_t)b * S_ * D_ + h * HD_;

    // Q fragments (16x32 per hd-chunk) straight from global
    v16bf aq[2];
#pragma unroll
    for (int c = 0; c < 2; c++) {
        const __bf16* qr = Qb + (size_t)(qrow0 + lh) * D_ + c * 32 + base;
        V16U u;
        u.h[0] = *(const v8bf*)qr;
        u.h[1] = *(const v8bf*)(qr + 16);
        aq[c] = u.v;
    }

    float mst[8], lst[8];
#pragma unroll
    for (int r = 0; r < 8; r++) { mst[r] = -1e30f; lst[r] = 0.f; }
    v8f acc[4] = {};
    const float scale = 0.125f;   // 1/sqrt(64)

    for (int kc = 0; kc < S_; kc += 32) {
        __syncthreads();          // prior iteration's reads of Ksub/Vt complete
#pragma unroll
        for (int i = 0; i < 2; i++) {   // async-stage K chunk: 32x64 = 256 x 16B
            int flat = t * 2 + i;
            int row = flat >> 2, c0 = (flat & 3) * 8;
            async_copy16(lds_addr32(&Ksub[row][c0]),
                         Kb + (size_t)(kc + row) * D_ + c0);
        }
#pragma unroll
        for (int i = 0; i < 2; i++) {   // stage V transposed: Vt[hd][key]
            int flat = t * 2 + i;       // 0..255
            int key = flat >> 3, hd0 = (flat & 7) * 8;
            v8bf v8 = *(const v8bf*)(Vb + (size_t)(kc + key) * D_ + hd0);
#pragma unroll
            for (int j = 0; j < 8; j++) Vt[hd0 + j][key] = v8[j];
        }
        wait_async0();
        __syncthreads();

        // scores: two 16x16 C fragments (keys 0..15 / 16..31), contraction over hd=64
        v8f s0 = {}, s1 = {};
#pragma unroll
        for (int c = 0; c < 2; c++) {
            v16bf bk0 = load_frag(&Ksub[0][0], 72, lh,      c * 32 + base);
            v16bf bk1 = load_frag(&Ksub[0][0], 72, 16 + lh, c * 32 + base);
            s0 = wmma_bf16(aq[c], bk0, s0);
            s1 = wmma_bf16(aq[c], bk1, s1);
        }

        // online softmax across the 32-key chunk; rows live per (half-group, r)
#pragma unroll
        for (int r = 0; r < 8; r++) {
            float x0 = s0[r] * scale, x1 = s1[r] * scale;
            float mx = fmaxf(x0, x1);
            for (int o = 8; o >= 1; o >>= 1) mx = fmaxf(mx, __shfl_xor(mx, o, 32));
            float mnew = fmaxf(mst[r], mx);
            float corr = __expf(mst[r] - mnew);
            float p0 = __expf(x0 - mnew), p1 = __expf(x1 - mnew);
            float rs = p0 + p1;
            for (int o = 8; o >= 1; o >>= 1) rs += __shfl_xor(rs, o, 32);
            lst[r] = lst[r] * corr + rs;
            mst[r] = mnew;
#pragma unroll
            for (int f = 0; f < 4; f++) acc[f][r] *= corr;
            int m = mhalf + r;
            Pscr[wave][m][lh]      = (__bf16)p0;
            Pscr[wave][m][16 + lh] = (__bf16)p1;
        }
        asm volatile("s_wait_dscnt 0" ::: "memory");   // wave-local P store -> load

        // ctx(16x64) += P(16x32) @ V(32x64)
        v16bf ap = load_frag(&Pscr[wave][0][0], 40, lh, base);
#pragma unroll
        for (int f = 0; f < 4; f++) {
            v16bf bv = load_frag(&Vt[0][0], 40, f * 16 + lh, base);
            acc[f] = wmma_bf16(ap, bv, acc[f]);
        }
    }

#pragma unroll
    for (int r = 0; r < 8; r++) {
        float inv = 1.0f / lst[r];
        size_t row = (size_t)b * S_ + qg * 64 + wave * 16 + mhalf + r;
#pragma unroll
        for (int f = 0; f < 4; f++)
            Ctx[row * D_ + h * HD_ + f * 16 + lh] = (__bf16)(acc[f][r] * inv);
    }
}

// ---------------- host-side orchestration ----------------
extern "C" void kernel_launch(void* const* d_in, const int* in_sizes, int n_in,
                              void* d_out, int out_size, void* d_ws, size_t ws_size,
                              hipStream_t stream) {
    (void)in_sizes; (void)n_in; (void)out_size; (void)ws_size;
    const float* hidden = (const float*)d_in[0];
    const float* n1g = (const float*)d_in[1];
    const float* n1b = (const float*)d_in[2];
    const float* wq  = (const float*)d_in[3];
    const float* bq  = (const float*)d_in[4];
    const float* wk  = (const float*)d_in[5];
    const float* bk  = (const float*)d_in[6];
    const float* wv  = (const float*)d_in[7];
    const float* bv  = (const float*)d_in[8];
    const float* wo  = (const float*)d_in[9];
    const float* bo  = (const float*)d_in[10];
    const float* ls1 = (const float*)d_in[11];
    const float* n2g = (const float*)d_in[12];
    const float* n2b = (const float*)d_in[13];
    const float* w1  = (const float*)d_in[14];
    const float* b1  = (const float*)d_in[15];
    const float* w2  = (const float*)d_in[16];
    const float* b2  = (const float*)d_in[17];
    const float* ls2 = (const float*)d_in[18];

    char* ws = (char*)d_ws;
    size_t off = 0;
    auto alloc = [&](size_t bytes) -> void* {
        void* p = ws + off;
        off += (bytes + 255) & ~(size_t)255;
        return p;
    };
    float*  x1  = (float*) alloc((size_t)M_ * D_ * 4);   // residual after attention
    __bf16* xn  = (__bf16*)alloc((size_t)M_ * D_ * 2);   // LN1 out; reused for ctx
    __bf16* qb  = (__bf16*)alloc((size_t)M_ * D_ * 2);   // q; reused for yn
    __bf16* kb  = (__bf16*)alloc((size_t)M_ * D_ * 2);
    __bf16* vb  = (__bf16*)alloc((size_t)M_ * D_ * 2);
    __bf16* hb  = (__bf16*)alloc((size_t)M_ * DFF_ * 2); // MLP hidden
    __bf16* wqt = (__bf16*)alloc((size_t)D_ * D_ * 2);   // all weights pre-transposed [N,K]
    __bf16* wkt = (__bf16*)alloc((size_t)D_ * D_ * 2);
    __bf16* wvt = (__bf16*)alloc((size_t)D_ * D_ * 2);
    __bf16* wot = (__bf16*)alloc((size_t)D_ * D_ * 2);
    __bf16* w1t = (__bf16*)alloc((size_t)D_ * DFF_ * 2);
    __bf16* w2t = (__bf16*)alloc((size_t)DFF_ * D_ * 2);

    // weight transpose+convert: in [K,N] f32 -> out [N,K] bf16
    dim3 tDD(D_ / 32, D_ / 32);
    transpose_bf16_kernel<<<tDD, 256, 0, stream>>>(wq, wqt, D_, D_);
    transpose_bf16_kernel<<<tDD, 256, 0, stream>>>(wk, wkt, D_, D_);
    transpose_bf16_kernel<<<tDD, 256, 0, stream>>>(wv, wvt, D_, D_);
    transpose_bf16_kernel<<<tDD, 256, 0, stream>>>(wo, wot, D_, D_);
    transpose_bf16_kernel<<<dim3(DFF_ / 32, D_ / 32), 256, 0, stream>>>(w1, w1t, D_, DFF_);
    transpose_bf16_kernel<<<dim3(D_ / 32, DFF_ / 32), 256, 0, stream>>>(w2, w2t, DFF_, D_);

    // LN1
    ln_bf16_kernel<<<M_, 256, 0, stream>>>(hidden, n1g, n1b, xn);

    // QKV projections (mode 0 -> bf16)
    dim3 gDD(D_ / 128, M_ / 256);
    gemm_bf16_kernel<<<gDD, 256, 0, stream>>>(xn, wqt, bq, nullptr, nullptr, qb, M_, D_, D_, 0);
    gemm_bf16_kernel<<<gDD, 256, 0, stream>>>(xn, wkt, bk, nullptr, nullptr, kb, M_, D_, D_, 0);
    gemm_bf16_kernel<<<gDD, 256, 0, stream>>>(xn, wvt, bv, nullptr, nullptr, vb, M_, D_, D_, 0);

    // attention -> ctx (reuse xn buffer)
    attn_kernel<<<B_ * H_ * 9, 128, 0, stream>>>(qb, kb, vb, xn);

    // O-proj: x1 = (ctx @ wo + bo) * ls1 + hidden
    gemm_bf16_kernel<<<gDD, 256, 0, stream>>>(xn, wot, bo, ls1, hidden, x1, M_, D_, D_, 1);

    // LN2 -> yn (reuse qb)
    ln_bf16_kernel<<<M_, 256, 0, stream>>>(x1, n2g, n2b, qb);

    // MLP1: hb = gelu(yn @ w1 + b1)
    dim3 gDF(DFF_ / 128, M_ / 256);
    gemm_bf16_kernel<<<gDF, 256, 0, stream>>>(qb, w1t, b1, nullptr, nullptr, hb, M_, DFF_, D_, 2);

    // MLP2: out = (hb @ w2 + b2) * ls2 + x1
    gemm_bf16_kernel<<<gDD, 256, 0, stream>>>(hb, w2t, b2, ls2, x1, (float*)d_out, M_, D_, DFF_, 1);
}
